// SIGMA_8306466750792
// MI455X (gfx1250) — compile-verified, matrix operand order checked by reading, not run on hardware
//
#include <hip/hip_runtime.h>
#include <hip/hip_bf16.h>
#include <math.h>

typedef __attribute__((ext_vector_type(16))) __bf16 v16bf;
typedef __attribute__((ext_vector_type(8)))  float  v8f;

constexpr int Bb  = 32, Ls = 200, Hh = 128, Vv = 30000;
constexpr int DIc = 256, DSc = 32, DTRc = 8;
constexpr int NT  = Bb * Ls;               // 6400 tokens
constexpr int KCUT = Ls - 5;               // 195

static inline int cdiv(int a, int b) { return (a + b - 1) / b; }

// ---------------------------------------------------------------- embedding + LN
__global__ __launch_bounds__(128) void embed_ln_kernel(
    const int* __restrict__ seq, const float* __restrict__ emb,
    const float* __restrict__ g, const float* __restrict__ b,
    float* __restrict__ x)
{
    int row = blockIdx.x, j = threadIdx.x;
    int it  = seq[row];
    float v = emb[(size_t)it * Hh + j];
    __shared__ float red[128];
    red[j] = v; __syncthreads();
    for (int s = 64; s > 0; s >>= 1) { if (j < s) red[j] += red[j + s]; __syncthreads(); }
    float mu = red[0] * (1.f / 128.f); __syncthreads();
    float dv = v - mu;
    red[j] = dv * dv; __syncthreads();
    for (int s = 64; s > 0; s >>= 1) { if (j < s) red[j] += red[j + s]; __syncthreads(); }
    float var = red[0] * (1.f / 128.f);
    x[(size_t)row * Hh + j] = dv * rsqrtf(var + 1e-12f) * g[j] + b[j];
}

// ------------------------------------------------------- LN(a [+ res]) per token
__global__ __launch_bounds__(128) void ln_res_kernel(
    const float* __restrict__ a, const float* __restrict__ res,
    const float* __restrict__ g, const float* __restrict__ b,
    float* __restrict__ out)
{
    int row = blockIdx.x, j = threadIdx.x;
    float v = a[(size_t)row * Hh + j];
    if (res) v += res[(size_t)row * Hh + j];
    __shared__ float red[128];
    red[j] = v; __syncthreads();
    for (int s = 64; s > 0; s >>= 1) { if (j < s) red[j] += red[j + s]; __syncthreads(); }
    float mu = red[0] * (1.f / 128.f); __syncthreads();
    float dv = v - mu;
    red[j] = dv * dv; __syncthreads();
    for (int s = 64; s > 0; s >>= 1) { if (j < s) red[j] += red[j + s]; __syncthreads(); }
    float var = red[0] * (1.f / 128.f);
    out[(size_t)row * Hh + j] = dv * rsqrtf(var + 1e-12f) * g[j] + b[j];
}

// ----------------------------------------- generic Y = X @ W^T (+bias)(+res)(act)
// act: 0 none, 1 softplus, 2 exact gelu
__global__ void linear_kernel(
    const float* __restrict__ X, int ldx,
    const float* __restrict__ W, const float* __restrict__ bias,
    const float* __restrict__ res, float* __restrict__ Y,
    int rows, int N, int K, int act)
{
    int idx = blockIdx.x * blockDim.x + threadIdx.x;
    if (idx >= rows * N) return;
    int row = idx / N, n = idx - row * N;
    const float* x = X + (size_t)row * ldx;
    const float* w = W + (size_t)n * K;
    float acc = bias ? bias[n] : 0.f;
    for (int k = 0; k < K; ++k) acc = fmaf(x[k], w[k], acc);
    if (res) acc += res[(size_t)row * N + n];
    if (act == 1) acc = fmaxf(acc, 0.f) + log1pf(expf(-fabsf(acc)));          // softplus
    else if (act == 2) acc = 0.5f * acc * (1.f + erff(acc * 0.70710678118f)); // gelu
    Y[(size_t)row * N + n] = acc;
}

// ------------------------------------------------ dense conv k=3, pad=1 (g1 path)
__global__ void conv3_kernel(
    const float* __restrict__ x, const float* __restrict__ w,
    const float* __restrict__ bias, float* __restrict__ out)
{
    int idx = blockIdx.x * blockDim.x + threadIdx.x;
    if (idx >= NT * Hh) return;
    int o = idx & 127, t = (idx >> 7) % Ls, b = idx / (Ls * Hh);
    float acc = bias[o];
    for (int k = 0; k < 3; ++k) {
        int tt = t - 1 + k;
        if (tt < 0 || tt >= Ls) continue;
        const float* xr = x + ((size_t)b * Ls + tt) * Hh;
        const float* wr = w + (size_t)o * (Hh * 3) + k;   // [o][i][k]
        for (int i = 0; i < Hh; ++i) acc = fmaf(xr[i], wr[i * 3], acc);
    }
    out[idx] = acc;
}

// -------------------------------------- mamba depthwise causal conv k=4 + SiLU
__global__ void dwconv_silu_kernel(
    const float* __restrict__ xz, const float* __restrict__ cw,
    const float* __restrict__ cb, float* __restrict__ xc)
{
    int idx = blockIdx.x * blockDim.x + threadIdx.x;
    if (idx >= NT * DIc) return;
    int d = idx & 255, t = (idx >> 8) % Ls, b = idx / (Ls * DIc);
    float acc = cb[d];
    for (int k = 0; k < 4; ++k) {
        int tt = t - 3 + k;
        if (tt >= 0)
            acc = fmaf(xz[((size_t)b * Ls + tt) * (2 * DIc) + d], cw[d * 4 + k], acc);
    }
    xc[idx] = acc / (1.f + expf(-acc));   // silu
}

// ------------------------------------------------------- mamba selective scan
__global__ __launch_bounds__(256) void mamba_scan_kernel(
    const float* __restrict__ dtb, const float* __restrict__ xp,
    const float* __restrict__ xc,  const float* __restrict__ xz,
    const float* __restrict__ A_log, const float* __restrict__ Dp,
    float* __restrict__ yb)
{
    int idx = blockIdx.x * blockDim.x + threadIdx.x;   // B*DI = 8192
    int b = idx >> 8, d = idx & 255;
    float A[DSc], h[DSc];
#pragma unroll
    for (int s = 0; s < DSc; ++s) { A[s] = -expf(A_log[d * DSc + s]); h[s] = 0.f; }
    float Dv = Dp[d];
    for (int t = 0; t < Ls; ++t) {
        size_t rowt = (size_t)b * Ls + t;
        float dt = dtb[rowt * DIc + d];
        float xv = xc [rowt * DIc + d];
        const float* Bm = xp + rowt * 72 + DTRc;
        const float* Cm = Bm + DSc;
        float y = 0.f;
#pragma unroll
        for (int s = 0; s < DSc; ++s) {
            h[s] = expf(dt * A[s]) * h[s] + dt * Bm[s] * xv;
            y = fmaf(h[s], Cm[s], y);
        }
        y = fmaf(xv, Dv, y);
        float z = xz[rowt * (2 * DIc) + DIc + d];
        y *= z / (1.f + expf(-z));        // * silu(z)
        yb[rowt * DIc + d] = y;
    }
}

// ------------------------------------------------------------------------ GRU
__global__ __launch_bounds__(384) void gru_kernel(
    const float* __restrict__ gi, const float* __restrict__ whh,
    float* __restrict__ out)
{
    int b = blockIdx.x, j = threadIdx.x;   // j in [0,384)
    __shared__ float hs[128];
    __shared__ float ghs[384];
    if (j < 128) hs[j] = 0.f;
    __syncthreads();
    for (int t = 0; t < Ls; ++t) {
        const float* w = whh + (size_t)j * Hh;
        float acc = 0.f;
        for (int k = 0; k < Hh; ++k) acc = fmaf(hs[k], w[k], acc);
        ghs[j] = acc;
        __syncthreads();
        if (j < 128) {
            const float* g = gi + ((size_t)b * Ls + t) * 384;
            float r  = 1.f / (1.f + expf(-(g[j]       + ghs[j])));
            float zt = 1.f / (1.f + expf(-(g[128 + j] + ghs[128 + j])));
            float n  = tanhf(g[256 + j] + r * ghs[256 + j]);
            float hn = (1.f - zt) * n + zt * hs[j];
            out[((size_t)b * Ls + t) * Hh + j] = hn;
            hs[j] = hn;
        }
        __syncthreads();
    }
}

// ------------------------------------------------ gate(h) = silu(h)Wsi + sig(h)Wsg
__global__ void gate_kernel(
    const float* __restrict__ h,
    const float* __restrict__ wsi, const float* __restrict__ bsi,
    const float* __restrict__ wsg, const float* __restrict__ bsg,
    float* __restrict__ out)
{
    int idx = blockIdx.x * blockDim.x + threadIdx.x;
    if (idx >= NT * Hh) return;
    int row = idx >> 7, o = idx & 127;
    const float* hr = h   + (size_t)row * Hh;
    const float* w1 = wsi + (size_t)o * Hh;
    const float* w2 = wsg + (size_t)o * Hh;
    float a1 = bsi[o], a2 = bsg[o];
    for (int i = 0; i < Hh; ++i) {
        float hv = hr[i];
        float sg = 1.f / (1.f + expf(-hv));
        a1 = fmaf(hv * sg, w1[i], a1);
        a2 = fmaf(sg,      w2[i], a2);
    }
    out[idx] = a1 + a2;
}

// ----------------------------------------------------------------- elementwise
__global__ void mul_kernel(const float* a, const float* b, float* o, int n)
{ int i = blockIdx.x * blockDim.x + threadIdx.x; if (i < n) o[i] = a[i] * b[i]; }

__global__ void flip_kernel(const float* __restrict__ x, float* __restrict__ out)
{
    int idx = blockIdx.x * blockDim.x + threadIdx.x;
    if (idx >= NT * Hh) return;
    int j = idx & 127, t = (idx >> 7) % Ls, b = idx / (Ls * Hh);
    int ts = (t < KCUT) ? (KCUT - 1 - t) : t;
    out[idx] = x[((size_t)b * Ls + ts) * Hh + j];
}

__global__ void softmax3_kernel(const float* cw, float* w)
{
    float a = cw[0], b = cw[1], c = cw[2];
    float m = fmaxf(a, fmaxf(b, c));
    float ea = expf(a - m), eb = expf(b - m), ec = expf(c - m);
    float s = ea + eb + ec;
    w[0] = ea / s; w[1] = eb / s; w[2] = ec / s;
}

__global__ void combine_kernel(const float* m1, const float* m2, const float* g,
                               const float* __restrict__ w, float* o, int n)
{
    int i = blockIdx.x * blockDim.x + threadIdx.x;
    if (i < n) o[i] = w[2] * m1[i] + w[1] * m2[i] + w[0] * g[i];
}

__global__ void cvt_bf16_kernel(const float* __restrict__ in,
                                unsigned short* __restrict__ out, int n)
{
    int i = blockIdx.x * blockDim.x + threadIdx.x;
    if (i >= n) return;
    __bf16 b = (__bf16)in[i];
    out[i] = *(unsigned short*)&b;
}

// --------------------------------------------------- logits = seq @ emb^T (WMMA)
// A = seq [M,128] bf16 row-major, B^T = emb [N,128] bf16 row-major.
// 2x2 register-blocked: each wave computes a 32(M) x 32(N) block = 4 tiles,
// 16 v_wmma per wave, fragments reused across the pair -> half the L2 reads.
// Column-1 validity is a compile-time template parameter so the K-loop body is
// branch-free and every WMMA executes under untouched, all-1s EXEC.
__device__ __forceinline__ v8f wmma_bf16(const v16bf& a, const v16bf& b, const v8f& c)
{
    return __builtin_amdgcn_wmma_f32_16x16x32_bf16(false, a, false, b,
                                                   (short)0, c, false, false);
}

template <bool N1>
__device__ __forceinline__ void wmma_tilepair(
    const unsigned short* __restrict__ a0, const unsigned short* __restrict__ a1,
    const unsigned short* __restrict__ b0, const unsigned short* __restrict__ b1,
    int kh, v8f& c00, v8f& c01, v8f& c10, v8f& c11)
{
#pragma unroll
    for (int kk = 0; kk < Hh; kk += 32) {
        union { uint4 u[2]; v16bf v; } A0, A1, B0, B1;
        A0.u[0] = *(const uint4*)(a0 + kk +      kh * 8);
        A0.u[1] = *(const uint4*)(a0 + kk + 16 + kh * 8);
        A1.u[0] = *(const uint4*)(a1 + kk +      kh * 8);
        A1.u[1] = *(const uint4*)(a1 + kk + 16 + kh * 8);
        B0.u[0] = *(const uint4*)(b0 + kk +      kh * 8);
        B0.u[1] = *(const uint4*)(b0 + kk + 16 + kh * 8);
        if constexpr (N1) {
            B1.u[0] = *(const uint4*)(b1 + kk +      kh * 8);
            B1.u[1] = *(const uint4*)(b1 + kk + 16 + kh * 8);
        }
        c00 = wmma_bf16(A0.v, B0.v, c00);
        c10 = wmma_bf16(A1.v, B0.v, c10);
        if constexpr (N1) {
            c01 = wmma_bf16(A0.v, B1.v, c01);
            c11 = wmma_bf16(A1.v, B1.v, c11);
        }
    }
}

__global__ __launch_bounds__(256) void wmma_logits_kernel(
    const unsigned short* __restrict__ seqb,
    const unsigned short* __restrict__ embb,
    float* __restrict__ out, int M, int N)
{
    int wave = threadIdx.x >> 5;
    int lane = threadIdx.x & 31;
    int tn0  = (blockIdx.x * 8 + wave) * 2;   // N-tile pair base
    int tm0  = blockIdx.y * 2;                // M-tile pair base (M/16 is even)
    if (tn0 * 16 >= N) return;                // wave-uniform
    bool n1ok = (tn0 + 1) * 16 < N;           // wave-uniform

    int r  = lane & 15;                       // row within tile (A and B^T)
    int kh = lane >> 4;                       // K-half selector
    const unsigned short* a0 = seqb + (size_t)(tm0 * 16 + r) * Hh;
    const unsigned short* a1 = a0 + (size_t)16 * Hh;
    const unsigned short* b0 = embb + (size_t)(tn0 * 16 + r) * Hh;
    const unsigned short* b1 = b0 + (size_t)16 * Hh;

    v8f c00 = {}, c01 = {}, c10 = {}, c11 = {};
    if (n1ok) wmma_tilepair<true >(a0, a1, b0, b1, kh, c00, c01, c10, c11);
    else      wmma_tilepair<false>(a0, a1, b0, b1, kh, c00, c01, c10, c11);

    // C/D layout: lane -> column (lane&15), VGPR i -> row i + 8*(lane>=16)
    auto store_tile = [&](const v8f& c, int tm, int tn) {
        float* orow = out + (size_t)(tm * 16) * N + tn * 16;
#pragma unroll
        for (int i = 0; i < 8; ++i) {
            int m = i + kh * 8;
            orow[(size_t)m * N + r] = c[i];
        }
    };
    store_tile(c00, tm0,     tn0);
    store_tile(c10, tm0 + 1, tn0);
    if (n1ok) {
        store_tile(c01, tm0,     tn0 + 1);
        store_tile(c11, tm0 + 1, tn0 + 1);
    }
}

// ============================================================================
extern "C" void kernel_launch(void* const* d_in, const int* in_sizes, int n_in,
                              void* d_out, int out_size, void* d_ws, size_t ws_size,
                              hipStream_t stream)
{
    (void)in_sizes; (void)n_in; (void)out_size; (void)ws_size;
    const int*   item_seq = (const int*)  d_in[0];
    const float* item_emb = (const float*)d_in[1];
    const float* ln_g  = (const float*)d_in[2];  const float* ln_b  = (const float*)d_in[3];
    const float* cw    = (const float*)d_in[4];
    const float* d1_w  = (const float*)d_in[5];  const float* d1_b  = (const float*)d_in[6];
    const float* d2_w  = (const float*)d_in[7];  const float* d2_b  = (const float*)d_in[8];
    const float* proj_w= (const float*)d_in[9];  const float* proj_b= (const float*)d_in[10];
    const float* conv_w= (const float*)d_in[11]; const float* conv_b= (const float*)d_in[12];
    const float* gsig_w= (const float*)d_in[13]; const float* gsig_b= (const float*)d_in[14];
    const float* gsi_w = (const float*)d_in[15]; const float* gsi_b = (const float*)d_in[16];
    const float* gru_wih=(const float*)d_in[17]; const float* gru_whh=(const float*)d_in[18];
    const float* m_in_w =(const float*)d_in[19];
    const float* m_conv_w=(const float*)d_in[20];const float* m_conv_b=(const float*)d_in[21];
    const float* m_x_w  =(const float*)d_in[22];
    const float* m_dt_w =(const float*)d_in[23]; const float* m_dt_b =(const float*)d_in[24];
    const float* m_A_log=(const float*)d_in[25]; const float* m_D    =(const float*)d_in[26];
    const float* m_out_w=(const float*)d_in[27];
    const float* bln_g =(const float*)d_in[28];  const float* bln_b =(const float*)d_in[29];
    const float* w1_w  =(const float*)d_in[30];  const float* w1_b  =(const float*)d_in[31];
    const float* w2_w  =(const float*)d_in[32];  const float* w2_b  =(const float*)d_in[33];
    const float* fln_g =(const float*)d_in[34];  const float* fln_b =(const float*)d_in[35];

    float* seq_out = (float*)d_out;                       // [NT,128]
    float* logits  = (float*)d_out + (size_t)NT * Hh;     // [NT,30000]

    // scratch bump allocator (16B aligned slices)
    float* base = (float*)d_ws; size_t off = 0;
    auto alloc = [&](size_t n) { float* p = base + off; off += (n + 3) & ~(size_t)3; return p; };
    float* x    = alloc((size_t)NT * Hh);
    float* h1   = alloc((size_t)NT * Hh);
    float* g1   = alloc((size_t)NT * Hh);
    float* gruo = alloc((size_t)NT * Hh);
    float* flp  = alloc((size_t)NT * Hh);
    float* h2   = alloc((size_t)NT * Hh);
    float* xzb  = alloc((size_t)NT * 2 * DIc);   // shared: gi / xz / FFN hidden
    float* xcb  = alloc((size_t)NT * DIc);
    float* xpb  = alloc((size_t)NT * 72);
    float* dtb  = alloc((size_t)NT * DIc);
    float* ybb  = alloc((size_t)NT * DIc);
    float* mo   = alloc((size_t)NT * Hh);
    float* gbuf = alloc((size_t)NT * Hh);
    float* m1g  = alloc((size_t)NT * Hh);
    float* m2g  = alloc((size_t)NT * Hh);
    float* comb = alloc((size_t)NT * Hh);
    float* tmp  = alloc((size_t)NT * Hh);
    float* out1 = alloc((size_t)NT * Hh);
    float* w3   = alloc(4);
    unsigned short* emb_bf = (unsigned short*)alloc(((size_t)Vv * Hh + 1) / 2);
    unsigned short* seq_bf = (unsigned short*)alloc(((size_t)NT * Hh + 1) / 2);

    const int TB = 256;
    auto lin = [&](const float* X, int ldx, const float* W, const float* bias,
                   const float* res, float* Y, int N, int K, int act) {
        linear_kernel<<<cdiv(NT * N, TB), TB, 0, stream>>>(X, ldx, W, bias, res, Y, NT, N, K, act);
    };

    // ---- embedding + input LN
    embed_ln_kernel<<<NT, 128, 0, stream>>>(item_seq, item_emb, ln_g, ln_b, x);
    // ---- h1, g1, GRU, flipped, h2
    lin(x, Hh, d1_w, d1_b, nullptr, h1, Hh, Hh, 0);
    conv3_kernel<<<cdiv(NT * Hh, TB), TB, 0, stream>>>(x, conv_w, conv_b, g1);
    lin(g1, Hh, gru_wih, nullptr, nullptr, xzb, 3 * Hh, Hh, 0);   // gi (reuses xzb)
    gru_kernel<<<Bb, 384, 0, stream>>>(xzb, gru_whh, gruo);
    flip_kernel<<<cdiv(NT * Hh, TB), TB, 0, stream>>>(x, flp);
    lin(flp, Hh, d2_w, d2_b, flp, h2, Hh, Hh, 0);

    // ---- mamba (shared scratch, run twice)
    auto run_mamba = [&](const float* xin, float* mout) {
        lin(xin, Hh, m_in_w, nullptr, nullptr, xzb, 2 * DIc, Hh, 0);                 // xz
        dwconv_silu_kernel<<<cdiv(NT * DIc, TB), TB, 0, stream>>>(xzb, m_conv_w, m_conv_b, xcb);
        lin(xcb, DIc, m_x_w, nullptr, nullptr, xpb, 72, DIc, 0);                     // xp
        lin(xpb, 72, m_dt_w, m_dt_b, nullptr, dtb, DIc, DTRc, 1);                    // dt softplus
        mamba_scan_kernel<<<(Bb * DIc) / 256, 256, 0, stream>>>(dtb, xpb, xcb, xzb,
                                                                m_A_log, m_D, ybb);
        lin(ybb, DIc, m_out_w, nullptr, nullptr, mout, Hh, DIc, 0);                  // out_proj
    };

    run_mamba(x, mo);
    gate_kernel<<<cdiv(NT * Hh, TB), TB, 0, stream>>>(h1, gsi_w, gsi_b, gsig_w, gsig_b, gbuf);
    mul_kernel<<<cdiv(NT * Hh, TB), TB, 0, stream>>>(mo, gbuf, m1g, NT * Hh);

    run_mamba(flp, mo);
    gate_kernel<<<cdiv(NT * Hh, TB), TB, 0, stream>>>(h2, gsi_w, gsi_b, gsig_w, gsig_b, gbuf);
    mul_kernel<<<cdiv(NT * Hh, TB), TB, 0, stream>>>(mo, gbuf, m2g, NT * Hh);

    // ---- combine + proj + residual LN
    softmax3_kernel<<<1, 1, 0, stream>>>(cw, w3);
    combine_kernel<<<cdiv(NT * Hh, TB), TB, 0, stream>>>(m1g, m2g, gruo, w3, comb, NT * Hh);
    lin(comb, Hh, proj_w, proj_b, nullptr, tmp, Hh, Hh, 0);
    ln_res_kernel<<<NT, 128, 0, stream>>>(tmp, x, bln_g, bln_b, out1);

    // ---- FFN + final LN -> seq (first chunk of d_out)
    lin(out1, Hh, w1_w, w1_b, nullptr, xzb, 4 * Hh, Hh, 2);          // gelu, reuses xzb
    lin(xzb, 4 * Hh, w2_w, w2_b, nullptr, tmp, Hh, 4 * Hh, 0);
    ln_res_kernel<<<NT, 128, 0, stream>>>(tmp, out1, fln_g, fln_b, seq_out);

    // ---- bf16 convert + WMMA logits GEMM
    cvt_bf16_kernel<<<cdiv(Vv * Hh, TB), TB, 0, stream>>>(item_emb, emb_bf, Vv * Hh);
    cvt_bf16_kernel<<<cdiv(NT * Hh, TB), TB, 0, stream>>>(seq_out, seq_bf, NT * Hh);
    // N tiles = 1875; each block: 8 waves x 2 N-tiles = 16 N-tiles
    dim3 ggrid(cdiv(cdiv(Vv, 16), 16), (NT / 16) / 2);   // (118, 200)
    wmma_logits_kernel<<<ggrid, 256, 0, stream>>>(seq_bf, emb_bf, logits, NT, Vv);
}
